// SurfaceConstrainedUp_43568148251474
// MI455X (gfx1250) — matrix operand order repaired and checked by reading, not compiled, not run on hardware
//
#include <hip/hip_runtime.h>
#include <hip/hip_bf16.h>

#define BB 8
#define NN 2048
#define UU 4
#define NK 16

typedef __attribute__((ext_vector_type(16))) __bf16 bf16x16;
typedef __attribute__((ext_vector_type(8)))  __bf16 bf16x8;
typedef __attribute__((ext_vector_type(8)))  float  f32x8;

__device__ __forceinline__ unsigned short bf_bits(float f) {
  unsigned u = __builtin_bit_cast(unsigned, f);
  unsigned r = (u + 0x7FFFu + ((u >> 16) & 1u)) >> 16;
  return (unsigned short)r;
}
__device__ __forceinline__ bf16x16 cat8(bf16x8 lo, bf16x8 hi) {
  return __builtin_shufflevector(lo, hi, 0,1,2,3,4,5,6,7,8,9,10,11,12,13,14,15);
}

// ---------------- generic WMMA GEMM: Y[b,o,m] = act(W[o,:]·X[:,m] + bias + res) -----------
// Src.shift: column m maps to source column m>>shift (shift 0: dense, 2: repeat4, 30: broadcast)
struct Src {
  const float* p;
  int ch0, nch, ncols, shift;
};
struct GemmP {
  const float* W; int wo_stride, wc_stride, w_off;
  const float* bias;
  Src src[4]; int nsrc;
  Src res; int has_res;
  const float* zb;           // zero page for padded channels
  float* out; int out_cols, out_stride, out_off;
  int O, C, ncols, act;      // act: 0 none, 1 relu, 2 tanh
};

// Block: 256 threads = 8 waves. Tile: 16 (O) x 128 (cols). K staged in chunks of 32.
// LDS layouts chosen so each lane's WMMA fragment is contiguous (2x ds_load_b128):
//   As[o][kk]   stride 32 ushorts : A frag = As[row*32 + half*8 ..] and [.. +16+half*8 ..]
//   Bs[col][kk] stride 40 ushorts (80B, 16B-aligned, bank-spread): B frag = Bs[col*40 + half*8], [+16+half*8]
// Per-channel source descriptors resolved once into rowP/rowSh -> guard-free staging loads.
// A-stage uses index clamps (not branches): k>=C pairs with zeroed B columns, o>=O rows are
// never stored, so clamped garbage never reaches the output.
#define BSTR 40
#define CMAX 768
__global__ __launch_bounds__(256) void gemm_wmma(GemmP P) {
  __shared__ __align__(16) unsigned short As[16 * 32];
  __shared__ __align__(16) unsigned short Bs[128 * BSTR];
  __shared__ const float* rowP[CMAX];
  __shared__ int          rowSh[CMAX];
  int tid = threadIdx.x;
  int lane = tid & 31, wid = tid >> 5;
  int half = lane >> 4, l15 = lane & 15;
  int b = blockIdx.z;
  int oBase = blockIdx.x * 16;
  int colBlk = blockIdx.y * 128;
  int col = wid * 16 + l15;       // 0..127 within tile
  int m = colBlk + col;
  int Cpad = (P.C + 31) & ~31;
  // resolve per-channel source descriptors once
  for (int c = tid; c < Cpad; c += 256) {
    const float* p = P.zb;
    int sh = 30;
    if (c < P.C) {
      #pragma unroll
      for (int i = 0; i < 4; ++i) {
        if (i < P.nsrc) {
          const Src& s = P.src[i];
          if (c >= s.ch0 && c < s.ch0 + s.nch) {
            p = s.p + ((size_t)b * s.nch + (c - s.ch0)) * s.ncols;
            sh = s.shift;
          }
        }
      }
    }
    rowP[c] = p;
    rowSh[c] = sh;
  }
  __syncthreads();
  f32x8 acc = {0.f,0.f,0.f,0.f,0.f,0.f,0.f,0.f};
  for (int kc = 0; kc < P.C; kc += 32) {
    // stage A (16x32): 512 elems, 2 per thread; clamped (branch-free) loads
    #pragma unroll
    for (int i = 0; i < 2; ++i) {
      int e = tid + i * 256;
      int o = e >> 5, kk = e & 31;
      int orow = min(oBase + o, P.O - 1);
      int ka = min(kc + kk, P.C - 1);
      float wv = P.W[(size_t)orow * P.wo_stride + (size_t)ka * P.wc_stride + P.w_off];
      As[o * 32 + kk] = bf_bits(wv);
    }
    // stage B (32x128): 4096 elems, 16 per thread; guard-free via rowP/zero-page
    #pragma unroll
    for (int i = 0; i < 16; ++i) {
      int e = tid + i * 256;
      int cc = kc + (e >> 7);     // global channel
      int mm = e & 127;           // 0..127
      int gm = colBlk + mm;
      float x = rowP[cc][gm >> rowSh[cc]];
      Bs[mm * BSTR + (e >> 7)] = bf_bits(x);
    }
    __syncthreads();
    bf16x8 a0 = *(const bf16x8*)&As[l15 * 32 + half * 8];
    bf16x8 a1 = *(const bf16x8*)&As[l15 * 32 + 16 + half * 8];
    bf16x8 b0 = *(const bf16x8*)&Bs[col * BSTR + half * 8];
    bf16x8 b1 = *(const bf16x8*)&Bs[col * BSTR + 16 + half * 8];
    acc = __builtin_amdgcn_wmma_f32_16x16x32_bf16(false, cat8(a0, a1), false, cat8(b0, b1),
                                                  (short)0, acc, false, false);
    __syncthreads();
  }
  #pragma unroll
  for (int r = 0; r < 8; ++r) {
    int o = oBase + r + half * 8;
    if (o < P.O) {
      float y = acc[r];
      if (P.bias) y += P.bias[o];
      if (P.has_res) y += P.res.p[((size_t)b * P.res.nch + o) * P.res.ncols + (m >> P.res.shift)];
      if (P.act == 1) y = fmaxf(y, 0.f);
      else if (P.act == 2) y = tanhf(y);
      P.out[((size_t)b * P.O + o) * P.out_cols + (size_t)m * P.out_stride + P.out_off] = y;
    }
  }
}

// ---------------- zero page ----------------
__global__ void zerok(float* __restrict__ p, int n) {
  int i = blockIdx.x * 64 + threadIdx.x;
  if (i < n) p[i] = 0.f;
}

// ---------------- max over N per (b, c) ----------------
__global__ void maxred(const float* __restrict__ fea, float* __restrict__ mx) {
  __shared__ float sm[256];
  int bc = blockIdx.x;
  float v = -3.4e38f;
  for (int n = threadIdx.x; n < NN; n += 256) v = fmaxf(v, fea[(size_t)bc * NN + n]);
  sm[threadIdx.x] = v;
  __syncthreads();
  for (int s = 128; s > 0; s >>= 1) {
    if (threadIdx.x < s) sm[threadIdx.x] = fmaxf(sm[threadIdx.x], sm[threadIdx.x + s]);
    __syncthreads();
  }
  if (threadIdx.x == 0) mx[bc] = sm[0];
}

// ---------------- kNN top-16 by squared distance ----------------
__global__ void knn_topk(const float* __restrict__ pos, int* __restrict__ idx) {
  int gid = blockIdx.x * 256 + threadIdx.x; // b*N + n
  int b = gid / NN, n = gid % NN;
  const float* P = pos + (size_t)b * 3 * NN;
  float px = P[n], py = P[NN + n], pz = P[2 * NN + n];
  float bd[NK]; int bi[NK];
  #pragma unroll
  for (int t = 0; t < NK; ++t) { bd[t] = 3.4e38f; bi[t] = 0; }
  for (int j = 0; j < NN; ++j) {
    float dx = px - P[j], dy = py - P[NN + j], dz = pz - P[2 * NN + j];
    float d = dx * dx + dy * dy + dz * dz;
    if (d < bd[NK - 1]) {
      #pragma unroll
      for (int t = NK - 1; t >= 1; --t) {
        if (d < bd[t - 1])      { bd[t] = bd[t - 1]; bi[t] = bi[t - 1]; }
        else if (d < bd[t])     { bd[t] = d; bi[t] = j; }
      }
      if (d < bd[0]) { bd[0] = d; bi[0] = j; }
    }
  }
  #pragma unroll
  for (int t = 0; t < NK; ++t) idx[(size_t)gid * NK + t] = bi[t];
}

// ---------------- pos_rel -> pos1(64x3) -> bn -> relu ----------------
__global__ void pospe(const float* __restrict__ pos, const int* __restrict__ idx,
                      const float* __restrict__ Wp, const float* __restrict__ bp,
                      const float* __restrict__ g, const float* __restrict__ s,
                      float* __restrict__ hpe) {
  int b = blockIdx.y;
  int m = blockIdx.x * 256 + threadIdx.x; // over N*16
  int M16 = NN * NK;
  if (m >= M16) return;
  int n = m >> 4, k = m & 15;
  int j = idx[((size_t)b * NN + n) * NK + k];
  const float* P = pos + (size_t)b * 3 * NN;
  float d0 = P[n] - P[j];
  float d1 = P[NN + n] - P[NN + j];
  float d2 = P[2 * NN + n] - P[2 * NN + j];
  for (int c = 0; c < 64; ++c) {
    float h = Wp[c * 3] * d0 + Wp[c * 3 + 1] * d1 + Wp[c * 3 + 2] * d2 + bp[c];
    h = fmaxf(h * g[c] + s[c], 0.f);
    hpe[((size_t)b * 64 + c) * M16 + m] = h;
  }
}

// ---------------- fused attention: attn1->bn->relu->attn2->softmax->sum(attn*(v+pe)) ------
// W1/W2 staged to LDS bf16 once (rolled loops); all fragments are contiguous ds_load_b128 pairs.
//   Ws1[o][k]  stride 72 ushorts (144B, 16B-aligned)
//   Ws2[o][k]  stride 264 ushorts (528B, 16B-aligned)
//   Xt[col][c] stride 72   (col = neighbor index)
//   Ht[col][o] stride 264
#define W1STR 72
#define W2STR 264
__global__ __launch_bounds__(256) void attn_fused(
    const float* __restrict__ qry, const float* __restrict__ keyb,
    const float* __restrict__ valb, const float* __restrict__ pe,
    const int* __restrict__ idx,
    const float* __restrict__ W1, const float* __restrict__ b1,
    const float* __restrict__ g1, const float* __restrict__ s1,
    const float* __restrict__ W2, const float* __restrict__ b2,
    float* __restrict__ agg) {
  __shared__ __align__(16) unsigned short Ws1[256 * W1STR];
  __shared__ __align__(16) unsigned short Ws2[64 * W2STR];
  __shared__ __align__(16) unsigned short Xt[16 * W1STR];
  __shared__ __align__(16) unsigned short Ht[16 * W2STR];
  __shared__ float At[64 * 16];
  __shared__ float Vt[64 * 16];
  __shared__ int nidx[NK];
  const int N = NN, M16 = NN * NK;
  int b = blockIdx.y, n = blockIdx.x;
  int tid = threadIdx.x;
  if (tid < NK) nidx[tid] = idx[((size_t)b * N + n) * NK + tid];
  // stage weights (rolled: small static footprint)
  for (int e = tid; e < 256 * 64; e += 256)
    Ws1[(e >> 6) * W1STR + (e & 63)] = bf_bits(W1[e]);
  for (int e = tid; e < 64 * 256; e += 256)
    Ws2[(e >> 8) * W2STR + (e & 255)] = bf_bits(W2[e]);
  __syncthreads();
  for (int e = tid; e < 64 * 16; e += 256) {
    int c = e >> 4, k = e & 15;
    size_t base = (size_t)b * 64 + c;
    float p = pe[base * M16 + (size_t)n * 16 + k];
    Xt[k * W1STR + c] = bf_bits(qry[base * N + n] - keyb[base * N + nidx[k]] + p);
    Vt[c * 16 + k] = valb[base * N + n] + p;
  }
  __syncthreads();
  int lane = tid & 31, wid = tid >> 5, half = lane >> 4, l15 = lane & 15;
  // stage 1: H(256x16) = relu(bn(W1(256x64) * X(64x16) + b1))
  #pragma unroll
  for (int t = 0; t < 2; ++t) {
    int oB = (wid + t * 8) * 16;
    int orow = oB + l15;
    f32x8 acc = {0.f,0.f,0.f,0.f,0.f,0.f,0.f,0.f};
    #pragma unroll
    for (int kc = 0; kc < 64; kc += 32) {
      bf16x8 a0 = *(const bf16x8*)&Ws1[orow * W1STR + kc + half * 8];
      bf16x8 a1 = *(const bf16x8*)&Ws1[orow * W1STR + kc + 16 + half * 8];
      bf16x8 b0 = *(const bf16x8*)&Xt[l15 * W1STR + kc + half * 8];
      bf16x8 b1 = *(const bf16x8*)&Xt[l15 * W1STR + kc + 16 + half * 8];
      acc = __builtin_amdgcn_wmma_f32_16x16x32_bf16(false, cat8(a0, a1), false, cat8(b0, b1),
                                                    (short)0, acc, false, false);
    }
    #pragma unroll
    for (int r = 0; r < 8; ++r) {
      int o = oB + r + half * 8;
      float h = acc[r] + b1[o];
      h = fmaxf(h * g1[o] + s1[o], 0.f);
      Ht[l15 * W2STR + o] = bf_bits(h);
    }
  }
  __syncthreads();
  // stage 2: attn(64x16) = W2(64x256) * H + b2
  if (wid < 4) {
    int oB = wid * 16;
    int orow = oB + l15;
    f32x8 acc = {0.f,0.f,0.f,0.f,0.f,0.f,0.f,0.f};
    #pragma unroll
    for (int kc = 0; kc < 256; kc += 32) {
      bf16x8 a0 = *(const bf16x8*)&Ws2[orow * W2STR + kc + half * 8];
      bf16x8 a1 = *(const bf16x8*)&Ws2[orow * W2STR + kc + 16 + half * 8];
      bf16x8 b0 = *(const bf16x8*)&Ht[l15 * W2STR + kc + half * 8];
      bf16x8 b1 = *(const bf16x8*)&Ht[l15 * W2STR + kc + 16 + half * 8];
      acc = __builtin_amdgcn_wmma_f32_16x16x32_bf16(false, cat8(a0, a1), false, cat8(b0, b1),
                                                    (short)0, acc, false, false);
    }
    #pragma unroll
    for (int r = 0; r < 8; ++r) {
      int o = oB + r + half * 8;
      At[o * 16 + l15] = acc[r] + b2[o];
    }
  }
  __syncthreads();
  if (tid < 64) {
    float mmax = -3.4e38f;
    #pragma unroll
    for (int k = 0; k < 16; ++k) mmax = fmaxf(mmax, At[tid * 16 + k]);
    float e[16], ssum = 0.f;
    #pragma unroll
    for (int k = 0; k < 16; ++k) { e[k] = expf(At[tid * 16 + k] - mmax); ssum += e[k]; }
    float inv = 1.f / ssum, a = 0.f;
    #pragma unroll
    for (int k = 0; k < 16; ++k) a += e[k] * inv * Vt[tid * 16 + k];
    agg[((size_t)b * 64 + tid) * N + n] = a;
  }
}

// ---------------- elementwise helpers ----------------
__global__ void subk(const float* __restrict__ a, const float* __restrict__ b,
                     float* __restrict__ c, size_t tot) {
  size_t i = (size_t)blockIdx.x * 256 + threadIdx.x;
  if (i < tot) c[i] = a[i] - b[i];
}

__global__ void repeat4(const float* __restrict__ src, float* __restrict__ dst,
                        size_t tot /* B*C*4N */) {
  size_t i = (size_t)blockIdx.x * 256 + threadIdx.x;
  if (i >= tot) return;
  int m = (int)(i % (4 * NN));
  size_t bc = i / (4 * NN);
  dst[i] = src[bc * NN + (m >> 2)];
}

// ---------------- final: rot6d -> R, pos_emb quirk, child_dw, child_pos ----------------
__global__ void finalk(const float* __restrict__ pos, const float* __restrict__ rot6,
                       const float* __restrict__ coefb, const float* __restrict__ uvb,
                       float* __restrict__ cpos, float* __restrict__ Rout,
                       float* __restrict__ dwout) {
  const int M4 = 4 * NN;
  const int Mtot = BB * M4;
  int i = blockIdx.x * 256 + threadIdx.x;
  if (i >= Mtot) return;
  int b = i / M4, m = i % M4, n = m >> 2, u = m & 3;
  // rot6d_to_matrix (rot identical across u)
  const float* rp = rot6 + (size_t)b * 6 * NN + n;
  float a0 = rp[0], a1 = rp[NN], a2 = rp[2 * NN], a3 = rp[3 * NN], a4 = rp[4 * NN], a5 = rp[5 * NN];
  float nx = fmaxf(sqrtf(a0 * a0 + a1 * a1 + a2 * a2), 1e-8f);
  float x0 = a0 / nx, x1 = a1 / nx, x2 = a2 / nx;
  float c0 = x1 * a5 - x2 * a4;
  float c1 = x2 * a3 - x0 * a5;
  float c2 = x0 * a4 - x1 * a3;
  float nz = fmaxf(sqrtf(c0 * c0 + c1 * c1 + c2 * c2), 1e-8f);
  float z0 = c0 / nz, z1 = c1 / nz, z2 = c2 / nz;
  float y0 = z1 * x2 - z2 * x1;
  float y1 = z2 * x0 - z0 * x2;
  float y2 = z0 * x1 - z1 * x0;
  if (u == 0) {
    float R9[9] = {x0, y0, z0, x1, y1, z1, x2, y2, z2};
    float* ro = Rout + ((size_t)b * NN + n) * 36;
    #pragma unroll
    for (int uu = 0; uu < 4; ++uu)
      #pragma unroll
      for (int t = 0; t < 9; ++t) ro[uu * 9 + t] = R9[t];
  }
  // pos_emb reproduces the reference's (4,M).reshape(-1,4) index crossing exactly
  float xx[4], yy[4];
  #pragma unroll
  for (int j = 0; j < 4; ++j) {
    int f = 4 * i + j;
    int p = f / Mtot;
    int sidx = f - p * Mtot;
    int sb = sidx / M4, sm = sidx % M4;
    float xv = uvb[((size_t)sb * 2) * M4 + sm];
    float yv = uvb[((size_t)sb * 2 + 1) * M4 + sm];
    xx[j] = (p == 0) ? 0.f : (p == 1) ? xv : (p == 2) ? xv * xv : xv * xv * xv;
    yy[j] = (p == 0) ? 0.f : (p == 1) ? yv : (p == 2) ? yv * yv : yv * yv * yv;
  }
  float dw = 0.f;
  #pragma unroll
  for (int ax = 0; ax < 4; ++ax)
    #pragma unroll
    for (int ay = 0; ay < 4; ++ay) {
      float pf = xx[ax] * yy[ay] + 1e-4f;
      dw += coefb[((size_t)b * 16 + (ax * 4 + ay)) * NN + n] * pf;
    }
  float du = uvb[((size_t)b * 2) * M4 + m];
  float dv = uvb[((size_t)b * 2 + 1) * M4 + m];
  float d0 = x0 * du + y0 * dv + z0 * dw;
  float d1 = x1 * du + y1 * dv + z1 * dw;
  float d2 = x2 * du + y2 * dv + z2 * dw;
  const float* pp = pos + (size_t)b * 3 * NN + n;
  cpos[((size_t)b * 3 + 0) * M4 + m] = d0 + pp[0];
  cpos[((size_t)b * 3 + 1) * M4 + m] = d1 + pp[NN];
  cpos[((size_t)b * 3 + 2) * M4 + m] = d2 + pp[2 * NN];
  dwout[i] = dw;
}

// ---------------- host ----------------
static inline Src S(const float* p, int ch0, int nch, int ncols, int shift) {
  Src s; s.p = p; s.ch0 = ch0; s.nch = nch; s.ncols = ncols; s.shift = shift; return s;
}
static GemmP mk(const float* W, int O, int C, const float* bias, int ncols, float* out, int act,
                const float* zb) {
  GemmP P{};
  P.W = W; P.wo_stride = C; P.wc_stride = 1; P.w_off = 0;
  P.bias = bias; P.nsrc = 0; P.has_res = 0; P.zb = zb;
  P.out = out; P.out_cols = ncols; P.out_stride = 1; P.out_off = 0;
  P.O = O; P.C = C; P.ncols = ncols; P.act = act;
  return P;
}
static void run_gemm(GemmP& P, hipStream_t st) {
  dim3 g((P.O + 15) / 16, (P.ncols + 127) / 128, BB);
  gemm_wmma<<<g, 256, 0, st>>>(P);
}

extern "C" void kernel_launch(void* const* d_in, const int* in_sizes, int n_in,
                              void* d_out, int out_size, void* d_ws, size_t ws_size,
                              hipStream_t stream) {
  const float* gsf = (const float*)d_in[0];
  const float* pos = (const float*)d_in[1];
  const float* fea = (const float*)d_in[2];
  const float* man = (const float*)d_in[3];
  const float* coef_W0 = (const float*)d_in[4];  const float* coef_b0 = (const float*)d_in[5];
  const float* coef_W1 = (const float*)d_in[6];  const float* coef_b1 = (const float*)d_in[7];
  const float* dxy_W0  = (const float*)d_in[8];  const float* dxy_b0  = (const float*)d_in[9];
  const float* dxy_W1  = (const float*)d_in[10]; const float* dxy_b1  = (const float*)d_in[11];
  const float* mfd_W0  = (const float*)d_in[12]; const float* mfd_b0  = (const float*)d_in[13];
  const float* mfd_W1  = (const float*)d_in[14]; const float* mfd_b1  = (const float*)d_in[15];
  const float* m2_W0   = (const float*)d_in[16]; const float* m2_b0   = (const float*)d_in[17];
  const float* m2_W1   = (const float*)d_in[18]; const float* m2_b1   = (const float*)d_in[19];
  const float* ch_W0   = (const float*)d_in[20]; const float* ch_b0   = (const float*)d_in[21];
  const float* ch_W1   = (const float*)d_in[22]; const float* ch_b1   = (const float*)d_in[23];
  const float* ps_W0   = (const float*)d_in[24]; const float* ps_b0   = (const float*)d_in[25];
  const float* ps_W1   = (const float*)d_in[26]; const float* ps_b1   = (const float*)d_in[27];
  const float* ps_w    = (const float*)d_in[28];
  const float* rot_W0  = (const float*)d_in[29]; const float* rot_b0  = (const float*)d_in[30];
  const float* rot_W1  = (const float*)d_in[31]; const float* rot_b1  = (const float*)d_in[32];
  const float* a1_W    = (const float*)d_in[33]; const float* a1_b    = (const float*)d_in[34];
  const float* a2_W    = (const float*)d_in[35]; const float* a2_b    = (const float*)d_in[36];
  const float* abn_g   = (const float*)d_in[37]; const float* abn_b   = (const float*)d_in[38];
  const float* k_W     = (const float*)d_in[39]; const float* k_b     = (const float*)d_in[40];
  const float* le_W    = (const float*)d_in[41]; const float* le_b    = (const float*)d_in[42];
  const float* ls_W    = (const float*)d_in[43]; const float* ls_b    = (const float*)d_in[44];
  const float* p1_W    = (const float*)d_in[45]; const float* p1_b    = (const float*)d_in[46];
  const float* p2_W    = (const float*)d_in[47]; const float* p2_b    = (const float*)d_in[48];
  const float* pbn_g   = (const float*)d_in[49]; const float* pbn_b   = (const float*)d_in[50];
  const float* q_W     = (const float*)d_in[51]; const float* q_b     = (const float*)d_in[52];
  const float* v_W     = (const float*)d_in[53]; const float* v_b     = (const float*)d_in[54];

  const int N = NN, M4 = 4 * NN, M16 = NN * NK;
  float* w = (float*)d_ws;
  size_t off = 0;
  auto A = [&](size_t n) { float* p = w + off; off += n; return p; };
  float* zb    = A(64);
  float* mx    = A((size_t)BB * 128);
  float* h512  = A((size_t)BB * 512 * N);
  float* xpf   = A((size_t)BB * 128 * N);
  float* x64   = A((size_t)BB * 64 * N);
  float* keyb  = A((size_t)BB * 64 * N);
  float* qry   = A((size_t)BB * 64 * N);
  float* valb  = A((size_t)BB * 64 * N);
  int*   idx   = (int*)A((size_t)BB * N * NK);
  float* hpe   = A((size_t)BB * 64 * M16);
  float* pe    = A((size_t)BB * 64 * M16);
  float* agg   = A((size_t)BB * 64 * N);
  float* rel   = A((size_t)BB * 128 * N);
  float* relmx = A((size_t)BB * 128 * N);
  float* h256m = A((size_t)BB * 256 * N);
  float* mf    = A((size_t)BB * 64 * N);
  float* h16r  = A((size_t)BB * 16 * N);
  float* rot6  = A((size_t)BB * 6 * N);
  float* h16c  = A((size_t)BB * 16 * N);
  float* coefb = A((size_t)BB * 16 * N);
  float* hps   = A((size_t)BB * 256 * N);
  float* dmf   = A((size_t)BB * 64 * N);
  float* dmch  = A((size_t)BB * 128 * M4);
  float* hch   = A((size_t)BB * 128 * M4);
  float* hxy   = A((size_t)BB * 32 * M4);
  float* uvb   = A((size_t)BB * 2 * M4);

  float* out    = (float*)d_out;
  float* o_cpos = out;
  float* o_cfea = o_cpos + (size_t)BB * 3 * M4;
  float* o_cman = o_cfea + (size_t)BB * 128 * M4;
  float* o_R    = o_cman + (size_t)BB * 64 * M4;
  float* o_dw   = o_R + (size_t)BB * N * 36;

  // 0) zero page for padded channels
  zerok<<<dim3(1), 64, 0, stream>>>(zb, 64);

  // 1) channel-max of fea, and kNN indices
  maxred<<<dim3(BB * 128), 256, 0, stream>>>(fea, mx);
  knn_topk<<<dim3(BB * N / 256), 256, 0, stream>>>(pos, idx);

  // 2) mlp2: [fea | max(fea) | gsf] (768) -> 512 relu -> 128 (+fea)
  {
    GemmP P = mk(m2_W0, 512, 768, m2_b0, N, h512, 1, zb);
    P.nsrc = 3;
    P.src[0] = S(fea, 0, 128, N, 0);
    P.src[1] = S(mx, 128, 128, 1, 30);
    P.src[2] = S(gsf, 256, 512, 1, 30);
    run_gemm(P, stream);
  }
  {
    GemmP P = mk(m2_W1, 128, 512, m2_b1, N, xpf, 0, zb);
    P.nsrc = 1; P.src[0] = S(h512, 0, 512, N, 0);
    P.has_res = 1; P.res = S(fea, 0, 128, N, 0);
    run_gemm(P, stream);
  }

  // 3) transformer front: ls, k, q, v
  { GemmP P = mk(ls_W, 64, 128, ls_b, N, x64, 0, zb); P.nsrc = 1; P.src[0] = S(xpf, 0, 128, N, 0); run_gemm(P, stream); }
  { GemmP P = mk(k_W, 64, 64, k_b, N, keyb, 0, zb);  P.nsrc = 1; P.src[0] = S(x64, 0, 64, N, 0); run_gemm(P, stream); }
  { GemmP P = mk(q_W, 64, 64, q_b, N, qry, 0, zb);   P.nsrc = 1; P.src[0] = S(x64, 0, 64, N, 0); run_gemm(P, stream); }
  { GemmP P = mk(v_W, 64, 64, v_b, N, valb, 0, zb);  P.nsrc = 1; P.src[0] = S(x64, 0, 64, N, 0); run_gemm(P, stream); }

  // 4) pe = pos2(relu(bn(pos1(pos_rel))))
  pospe<<<dim3((M16 + 255) / 256, BB), 256, 0, stream>>>(pos, idx, p1_W, p1_b, pbn_g, pbn_b, hpe);
  { GemmP P = mk(p2_W, 64, 64, p2_b, M16, pe, 0, zb); P.nsrc = 1; P.src[0] = S(hpe, 0, 64, M16, 0); run_gemm(P, stream); }

  // 5) fused attention core -> agg
  attn_fused<<<dim3(N, BB), 256, 0, stream>>>(qry, keyb, valb, pe, idx,
                                              a1_W, a1_b, abn_g, abn_b, a2_W, a2_b, agg);

  // 6) rel = le(agg) + xpf
  { GemmP P = mk(le_W, 128, 64, le_b, N, rel, 0, zb); P.nsrc = 1; P.src[0] = S(agg, 0, 64, N, 0);
    P.has_res = 1; P.res = S(xpf, 0, 128, N, 0); run_gemm(P, stream); }
  subk<<<dim3((unsigned)(((size_t)BB * 128 * N + 255) / 256)), 256, 0, stream>>>(rel, xpf, relmx, (size_t)BB * 128 * N);

  // 7) manifold: [pos | man | rel | gsf] (707) -> 256 relu -> 64 (+man)
  {
    GemmP P = mk(mfd_W0, 256, 707, mfd_b0, N, h256m, 1, zb);
    P.nsrc = 4;
    P.src[0] = S(pos, 0, 3, N, 0);
    P.src[1] = S(man, 3, 64, N, 0);
    P.src[2] = S(rel, 67, 128, N, 0);
    P.src[3] = S(gsf, 195, 512, 1, 30);
    run_gemm(P, stream);
  }
  { GemmP P = mk(mfd_W1, 64, 256, mfd_b1, N, mf, 0, zb); P.nsrc = 1; P.src[0] = S(h256m, 0, 256, N, 0);
    P.has_res = 1; P.res = S(man, 0, 64, N, 0); run_gemm(P, stream); }

  // 8) rot / coef heads
  { GemmP P = mk(rot_W0, 16, 64, rot_b0, N, h16r, 1, zb); P.nsrc = 1; P.src[0] = S(mf, 0, 64, N, 0); run_gemm(P, stream); }
  { GemmP P = mk(rot_W1, 6, 16, rot_b1, N, rot6, 0, zb);  P.nsrc = 1; P.src[0] = S(h16r, 0, 16, N, 0); run_gemm(P, stream); }
  { GemmP P = mk(coef_W0, 16, 64, coef_b0, N, h16c, 1, zb); P.nsrc = 1; P.src[0] = S(mf, 0, 64, N, 0); run_gemm(P, stream); }
  { GemmP P = mk(coef_W1, 16, 16, coef_b1, N, coefb, 0, zb); P.nsrc = 1; P.src[0] = S(h16c, 0, 16, N, 0); run_gemm(P, stream); }

  // 9) mlp_ps: [xpf | rel-xpf | mf] (320) -> 256 relu -> 64
  {
    GemmP P = mk(ps_W0, 256, 320, ps_b0, N, hps, 1, zb);
    P.nsrc = 3;
    P.src[0] = S(xpf, 0, 128, N, 0);
    P.src[1] = S(relmx, 128, 128, N, 0);
    P.src[2] = S(mf, 256, 64, N, 0);
    run_gemm(P, stream);
  }
  { GemmP P = mk(ps_W1, 64, 256, ps_b1, N, dmf, 0, zb); P.nsrc = 1; P.src[0] = S(hps, 0, 256, N, 0); run_gemm(P, stream); }

  // 10) dmchild: 4 strided-output GEMMs over ps_w[c][o][k]
  for (int k = 0; k < UU; ++k) {
    GemmP P = mk(ps_w, 128, 64, nullptr, N, dmch, 0, zb);
    P.wo_stride = UU; P.wc_stride = 128 * UU; P.w_off = k;
    P.nsrc = 1; P.src[0] = S(dmf, 0, 64, N, 0);
    P.out_cols = M4; P.out_stride = UU; P.out_off = k;
    run_gemm(P, stream);
  }

  // 11) mlp_ch: [dmchild | repeat(mf)] (192) -> 128 relu -> 128 (+repeat(xpf)) -> child_fea out
  {
    GemmP P = mk(ch_W0, 128, 192, ch_b0, M4, hch, 1, zb);
    P.nsrc = 2;
    P.src[0] = S(dmch, 0, 128, M4, 0);
    P.src[1] = S(mf, 128, 64, N, 2);
    run_gemm(P, stream);
  }
  {
    GemmP P = mk(ch_W1, 128, 128, ch_b1, M4, o_cfea, 0, zb);
    P.nsrc = 1; P.src[0] = S(hch, 0, 128, M4, 0);
    P.has_res = 1; P.res = S(xpf, 0, 128, N, 2);
    run_gemm(P, stream);
  }

  // 12) dxy: 128 -> 32 relu -> 2 tanh
  { GemmP P = mk(dxy_W0, 32, 128, dxy_b0, M4, hxy, 1, zb); P.nsrc = 1; P.src[0] = S(dmch, 0, 128, M4, 0); run_gemm(P, stream); }
  { GemmP P = mk(dxy_W1, 2, 32, dxy_b1, M4, uvb, 2, zb);   P.nsrc = 1; P.src[0] = S(hxy, 0, 32, M4, 0); run_gemm(P, stream); }

  // 13) childman output (repeat mf by U)
  repeat4<<<dim3((unsigned)(((size_t)BB * 64 * M4 + 255) / 256)), 256, 0, stream>>>(mf, o_cman, (size_t)BB * 64 * M4);

  // 14) final geometry: R, pos_emb, child_dw, child_pos
  finalk<<<dim3((BB * M4 + 255) / 256), 256, 0, stream>>>(pos, rot6, coefb, uvb, o_cpos, o_R, o_dw);
}
